// Attention_33157147525297
// MI455X (gfx1250) — matrix operand order repaired and checked by reading, not compiled, not run on hardware
//
#include <hip/hip_runtime.h>
#include <hip/hip_bf16.h>
#include <math.h>

// ---------------------------------------------------------------------------
// Types for CDNA5 WMMA (wave32): v_wmma_f32_16x16x32_bf16
// ---------------------------------------------------------------------------
typedef __bf16 bf16;
typedef __attribute__((ext_vector_type(16))) __bf16 v16bf;
typedef __attribute__((ext_vector_type(8)))  __bf16 v8bf;
typedef __attribute__((ext_vector_type(8)))  float  v8f;
typedef __attribute__((ext_vector_type(4)))  unsigned int u32x4;
typedef __attribute__((ext_vector_type(8)))  unsigned int u32x8;

#define BATCH    64
#define NTOK     196
#define NPAD     208      // 13 * 16
#define KPAD     224      // 7 * 32 (token dim padded for bf16 WMMA K-steps)
#define DIM      768
#define HEADS    12
#define HEAD_DIM 64
#define ROWS     (BATCH * NTOK)      // 12544 = 784 * 16
#define QKV_DIM  (3 * DIM)           // 2304  = 144 * 16

// LDS layout for the fused attention kernel (bytes)
#define S_OFF   0
#define S_BYTES (NPAD * NPAD * 4)            // 173056  S fp32 [208][208]
#define VQ_OFF  (S_OFF + S_BYTES)            // 173056  Q [208][64] bf16, later V^T [64][224] bf16
#define VQ_BYTES (HEAD_DIM * KPAD * 2)       // 28672
#define K_OFF   (VQ_OFF + VQ_BYTES)          // 201728  K [208][64] bf16
#define K_BYTES (NPAD * HEAD_DIM * 2)        // 26624
#define P_OFF   (K_OFF + K_BYTES)            // 228352  P [208][224] bf16
#define P_BYTES (NPAD * KPAD * 2)            // 93184
#define SMEM_BYTES (P_OFF + P_BYTES)         // 321536 <= 327680 (320KB/WGP on CDNA5)

// ---------------------------------------------------------------------------
// Fragment helpers, per ISA 7.12.2 (wave32 layouts).
// A (16x32, MxK) bf16: lane L holds row m=L&15; g=L>>4;
//   element e (0..15): k = kbase + g*8 + (e&7) + (e>>3)*16  -> two runs of 8.
// B (32x16, KxN) bf16: lane L holds col n=L&15; g=L>>4;
//   element e (0..15): k = kbase + g*8 + (e&7) + (e>>3)*16  -> two runs of 8.
// C/D (16x16 f32): VGPR r holds row m = r + 8*g, col n = L&15.
// ---------------------------------------------------------------------------
__device__ __forceinline__ v16bf ld_frag_a(const bf16* rowp, int g) {
  const bf16* p = rowp + g * 8;
  union { v16bf v; v8bf h[2]; } u;
  u.h[0] = *reinterpret_cast<const v8bf*>(p);
  u.h[1] = *reinterpret_cast<const v8bf*>(p + 16);
  return u.v;
}
__device__ __forceinline__ v16bf ld_frag_b(const bf16* rowp, int g) {
  const bf16* p = rowp + g * 8;
  union { v16bf v; v8bf h[2]; } u;
  u.h[0] = *reinterpret_cast<const v8bf*>(p);
  u.h[1] = *reinterpret_cast<const v8bf*>(p + 16);
  return u.v;
}
__device__ __forceinline__ v16bf frag_a(const bf16* base, int ld, int row, int kbase, int g) {
  return ld_frag_a(base + (size_t)row * ld + kbase, g);
}
__device__ __forceinline__ v16bf frag_b(const bf16* base, int ld, int row, int kbase, int g) {
  return ld_frag_b(base + (size_t)row * ld + kbase, g);
}
__device__ __forceinline__ v8f wmma_bf16(v16bf a, v16bf b, v8f c) {
  return __builtin_amdgcn_wmma_f32_16x16x32_bf16(false, a, false, b, (short)0, c, false, false);
}

// ---------------------------------------------------------------------------
// fp32 -> bf16 elementwise converter
// ---------------------------------------------------------------------------
__global__ __launch_bounds__(256) void cvt_f32_bf16(const float* __restrict__ in,
                                                    bf16* __restrict__ out, int n) {
  int i = blockIdx.x * 256 + threadIdx.x;
  if (i < n) out[i] = (bf16)in[i];
}

// ---------------------------------------------------------------------------
// QKV projection: C[12544,2304] = X[12544,768] @ Wqkv[2304,768]^T (bf16->bf16).
// 2x2 register tile per wave (32x32 output): A/B fragments reused twice each
// -> 2 x b128 loads per WMMA. The q third of the output (cols < 768) is
// pre-scaled by 2^-3 (the softmax scale, exact in bf16).
// ---------------------------------------------------------------------------
__global__ __launch_bounds__(256) void gemm_qkv(const bf16* __restrict__ A,
                                                const bf16* __restrict__ W,
                                                bf16* __restrict__ C) {
  const int wid  = blockIdx.x * 8 + (threadIdx.x >> 5);
  const int lane = threadIdx.x & 31;
  const int g    = lane >> 4;
  const int NG   = QKV_DIM / 32;           // 72 column groups
  const int mg   = wid / NG;               // 392 row groups
  const int ng   = wid % NG;
  const int n0   = ng * 32 + (lane & 15);
  const bf16* a0 = A + (size_t)(mg * 32 + (lane & 15)) * DIM;
  const bf16* a1 = a0 + (size_t)16 * DIM;
  const bf16* w0 = W + (size_t)n0 * DIM;
  const bf16* w1 = w0 + (size_t)16 * DIM;
  v8f acc00 = {}, acc01 = {}, acc10 = {}, acc11 = {};
  for (int kc = 0; kc < DIM; kc += 32) {
    v16bf fa0 = ld_frag_a(a0 + kc, g);
    v16bf fa1 = ld_frag_a(a1 + kc, g);
    v16bf fb0 = ld_frag_b(w0 + kc, g);
    v16bf fb1 = ld_frag_b(w1 + kc, g);
    acc00 = wmma_bf16(fa0, fb0, acc00);
    acc01 = wmma_bf16(fa0, fb1, acc01);
    acc10 = wmma_bf16(fa1, fb0, acc10);
    acc11 = wmma_bf16(fa1, fb1, acc11);
  }
  const float s0 = (n0 < DIM)      ? 0.125f : 1.0f;   // q columns pre-scaled
  const float s1 = (n0 + 16 < DIM) ? 0.125f : 1.0f;
  const int r0 = mg * 32 + 8 * g;
  for (int r = 0; r < 8; ++r) {
    C[(size_t)(r0 + r) * QKV_DIM + n0]           = (bf16)(acc00[r] * s0);
    C[(size_t)(r0 + r) * QKV_DIM + n0 + 16]      = (bf16)(acc01[r] * s1);
    C[(size_t)(r0 + 16 + r) * QKV_DIM + n0]      = (bf16)(acc10[r] * s0);
    C[(size_t)(r0 + 16 + r) * QKV_DIM + n0 + 16] = (bf16)(acc11[r] * s1);
  }
}

// ---------------------------------------------------------------------------
// Tensor Data Mover: DMA a [rows=196 x cols=64] bf16 tile (row stride 2304
// elements) from global memory into LDS at byte offset lds_off.
// D# per ISA 8.3/8.4: group0 = {count=1 | lds_addr | global_addr | type=2},
// group1 = {data_size=2B, tensor_dim0=64, tensor_dim1=196, tile 64x196,
//           tensor_dim0_stride=2304}.
// ---------------------------------------------------------------------------
__device__ __forceinline__ void tdm_load_tile(unsigned long long gaddr, unsigned lds_off) {
  u32x4 g0 = { 1u,
               lds_off,
               (unsigned)(gaddr & 0xFFFFFFFFu),
               ((unsigned)((gaddr >> 32) & 0x01FFFFFFu)) | 0x80000000u };  // type=2
  u32x8 g1 = { 0x00010000u,           // data_size = 1 (2 bytes)
               (unsigned)HEAD_DIM << 16,   // tensor_dim0[15:0]
               (unsigned)NTOK << 16,       // tensor_dim0[31:16]=0 | tensor_dim1[15:0]
               (unsigned)HEAD_DIM << 16,   // tensor_dim1[31:16]=0 | tile_dim0
               (unsigned)NTOK,             // tile_dim1 | tile_dim2=0
               (unsigned)QKV_DIM,          // tensor_dim0_stride[31:0]
               0u,                         // stride[47:32] | dim1_stride[15:0]
               0u };
  asm volatile("tensor_load_to_lds %0, %1" :: "s"(g0), "s"(g1) : "memory");
}

// ---------------------------------------------------------------------------
// Fused attention core: one workgroup per (b,h).
//   Q,K tiles staged by the Tensor Data Mover (async DMA, TENSORcnt)
//   S = q_scaled @ k^T            (WMMA, fp32 accum, into LDS)
//   mixed = 0.5*f0*S[m] + f1/(deg+1) * sum_{nbr} S[n]   (sparse graph mix)
//   P = softmax(mixed)            (bf16 into LDS, padded cols zeroed)
//   O = P @ V                     (WMMA, V stored transposed in LDS)
// ---------------------------------------------------------------------------
__global__ __launch_bounds__(512, 1) void attn_core(const bf16* __restrict__ qkv,
                                                    const float* __restrict__ factors,
                                                    bf16* __restrict__ obuf) {
  extern __shared__ char smem[];
  float* S  = reinterpret_cast<float*>(smem + S_OFF);   // [208][208]
  bf16*  VQ = reinterpret_cast<bf16*>(smem + VQ_OFF);   // Q [208][64] then V^T [64][224]
  bf16*  Kl = reinterpret_cast<bf16*>(smem + K_OFF);    // K [208][64]
  bf16*  P  = reinterpret_cast<bf16*>(smem + P_OFF);    // [208][224]

  const int b = blockIdx.x / HEADS;
  const int h = blockIdx.x % HEADS;
  const int tid  = threadIdx.x;
  const int wid  = tid >> 5;
  const int lane = tid & 31;
  const int g    = lane >> 4;

  const bf16* base = qkv + (size_t)(b * NTOK) * QKV_DIM + h * HEAD_DIM;

  // ---- Stage A: TDM DMA of Q and K tiles into LDS (wave 0 issues + waits),
  //      other waves zero the padded rows [196,208) concurrently.
  if (wid == 0) {
    const unsigned long long gq = (unsigned long long)(uintptr_t)base;
    tdm_load_tile(gq,                                  (unsigned)VQ_OFF);
    tdm_load_tile(gq + (unsigned long long)DIM * 2ull, (unsigned)K_OFF);
    __builtin_amdgcn_s_wait_tensorcnt(0);
  } else {
    for (int idx = tid - 32; idx < (NPAD - NTOK) * HEAD_DIM; idx += 480) {
      const int n = NTOK + idx / HEAD_DIM, d = idx % HEAD_DIM;
      VQ[n * HEAD_DIM + d] = (bf16)0.0f;
      Kl[n * HEAD_DIM + d] = (bf16)0.0f;
    }
  }
  __syncthreads();

  // ---- Stage S: S[208][208] = Q @ K^T via WMMA, 13x13 tiles, K=64
  for (int t = wid; t < 13 * 13; t += 16) {
    const int tm = t / 13, tn = t % 13;
    v8f acc = {};
    for (int kc = 0; kc < HEAD_DIM; kc += 32) {
      v16bf a  = frag_a(VQ, HEAD_DIM, tm * 16 + (lane & 15), kc, g);
      v16bf bf = frag_b(Kl, HEAD_DIM, tn * 16 + (lane & 15), kc, g);
      acc = wmma_bf16(a, bf, acc);
    }
    const int col   = tn * 16 + (lane & 15);
    const int rbase = tm * 16 + 8 * g;
    for (int r = 0; r < 8; ++r) S[(rbase + r) * NPAD + col] = acc[r];
  }
  __syncthreads();

  // ---- Stage B: load V transposed (Q region is dead now) + graph-mix + softmax
  for (int idx = tid; idx < HEAD_DIM * KPAD; idx += 512) {
    const int d = idx / KPAD, n = idx % KPAD;
    bf16 vv = (bf16)0.0f;
    if (n < NTOK) vv = base[(size_t)n * QKV_DIM + 2 * DIM + d];
    VQ[idx] = vv;                      // V^T[d][n]
  }

  const float f0 = factors[0];
  const float f1 = factors[1];
  for (int m = wid; m < NPAD; m += 16) {
    if (m >= NTOK) {                   // padded rows: P = 0
      for (int j = lane; j < KPAD; j += 32) P[m * KPAD + j] = (bf16)0.0f;
      continue;
    }
    const int gi = m / 14, gj = m % 14;
    int nb[4]; int deg = 0;
    if (gi > 0)  nb[deg++] = m - 14;
    if (gi < 13) nb[deg++] = m + 14;
    if (gj > 0)  nb[deg++] = m - 1;
    if (gj < 13) nb[deg++] = m + 1;
    const float coeff = f1 / (float)(deg + 1);
    const float diag  = 0.5f * f0;

    float mx = -3.0e38f;
    for (int j = lane; j < NTOK; j += 32) {
      float s = diag * S[m * NPAD + j];
      for (int q = 0; q < deg; ++q) s += coeff * S[nb[q] * NPAD + j];
      mx = fmaxf(mx, s);
    }
    for (int off = 16; off > 0; off >>= 1) mx = fmaxf(mx, __shfl_xor(mx, off, 32));

    float sum = 0.0f;
    for (int j = lane; j < NTOK; j += 32) {
      float s = diag * S[m * NPAD + j];
      for (int q = 0; q < deg; ++q) s += coeff * S[nb[q] * NPAD + j];
      const float e = __expf(s - mx);
      sum += e;
      P[m * KPAD + j] = (bf16)e;
    }
    for (int off = 16; off > 0; off >>= 1) sum += __shfl_xor(sum, off, 32);
    const float inv = 1.0f / sum;
    for (int j = lane; j < KPAD; j += 32) {
      if (j < NTOK) P[m * KPAD + j] = (bf16)((float)P[m * KPAD + j] * inv);
      else          P[m * KPAD + j] = (bf16)0.0f;
    }
  }
  __syncthreads();

  // ---- Stage PV: O[208][64] = P[208][224] @ V[224][64]  (B from V^T, contiguous)
  for (int t = wid; t < 13 * 4; t += 16) {
    const int tm = t / 4, tn = t % 4;
    v8f acc = {};
    for (int kc = 0; kc < KPAD; kc += 32) {
      v16bf a  = frag_a(P,  KPAD, tm * 16 + (lane & 15), kc, g);
      v16bf bf = frag_b(VQ, KPAD, tn * 16 + (lane & 15), kc, g);
      acc = wmma_bf16(a, bf, acc);
    }
    const int d     = tn * 16 + (lane & 15);
    const int rbase = tm * 16 + 8 * g;
    for (int r = 0; r < 8; ++r) {
      const int m = rbase + r;
      if (m < NTOK)
        obuf[(size_t)(b * NTOK + m) * DIM + h * HEAD_DIM + d] = (bf16)acc[r];
    }
  }
}

// ---------------------------------------------------------------------------
// Output projection: out[12544,768] = O[12544,768] @ Wp[768,768]^T + bias
// 2x2 register tile per wave, fp32 output.
// ---------------------------------------------------------------------------
__global__ __launch_bounds__(256) void gemm_proj(const bf16* __restrict__ A,
                                                 const bf16* __restrict__ W,
                                                 const float* __restrict__ bias,
                                                 float* __restrict__ C) {
  const int wid  = blockIdx.x * 8 + (threadIdx.x >> 5);
  const int lane = threadIdx.x & 31;
  const int g    = lane >> 4;
  const int NG   = DIM / 32;               // 24 column groups
  const int mg   = wid / NG;               // 392 row groups
  const int ng   = wid % NG;
  const int n0   = ng * 32 + (lane & 15);
  const bf16* a0 = A + (size_t)(mg * 32 + (lane & 15)) * DIM;
  const bf16* a1 = a0 + (size_t)16 * DIM;
  const bf16* w0 = W + (size_t)n0 * DIM;
  const bf16* w1 = w0 + (size_t)16 * DIM;
  v8f acc00 = {}, acc01 = {}, acc10 = {}, acc11 = {};
  for (int kc = 0; kc < DIM; kc += 32) {
    v16bf fa0 = ld_frag_a(a0 + kc, g);
    v16bf fa1 = ld_frag_a(a1 + kc, g);
    v16bf fb0 = ld_frag_b(w0 + kc, g);
    v16bf fb1 = ld_frag_b(w1 + kc, g);
    acc00 = wmma_bf16(fa0, fb0, acc00);
    acc01 = wmma_bf16(fa0, fb1, acc01);
    acc10 = wmma_bf16(fa1, fb0, acc10);
    acc11 = wmma_bf16(fa1, fb1, acc11);
  }
  const float b0 = bias[n0];
  const float b1 = bias[n0 + 16];
  const int r0 = mg * 32 + 8 * g;
  for (int r = 0; r < 8; ++r) {
    C[(size_t)(r0 + r) * DIM + n0]           = acc00[r] + b0;
    C[(size_t)(r0 + r) * DIM + n0 + 16]      = acc01[r] + b1;
    C[(size_t)(r0 + 16 + r) * DIM + n0]      = acc10[r] + b0;
    C[(size_t)(r0 + 16 + r) * DIM + n0 + 16] = acc11[r] + b1;
  }
}

// ---------------------------------------------------------------------------
// Host side
// ---------------------------------------------------------------------------
extern "C" void kernel_launch(void* const* d_in, const int* in_sizes, int n_in,
                              void* d_out, int out_size, void* d_ws, size_t ws_size,
                              hipStream_t stream) {
  (void)in_sizes; (void)n_in; (void)out_size; (void)ws_size;
  const float* x       = (const float*)d_in[0];   // [64,196,768]
  const float* w_qkv   = (const float*)d_in[1];   // [2304,768]
  const float* w_proj  = (const float*)d_in[2];   // [768,768]
  const float* b_proj  = (const float*)d_in[3];   // [768]
  const float* factors = (const float*)d_in[4];   // [3]
  float* out = (float*)d_out;

  char* ws = (char*)d_ws;
  bf16* x_bf     = (bf16*)(ws);                    // 19,267,584 B
  bf16* wqkv_bf  = (bf16*)(ws + 19267584);         //  3,538,944 B
  bf16* wproj_bf = (bf16*)(ws + 22806528);         //  1,179,648 B
  bf16* qkv_bf   = (bf16*)(ws + 23986176);         // 57,802,752 B
  bf16* o_bf     = (bf16*)(ws + 81788928);         // 19,267,584 B  (total ~101 MB)

  const int n_x  = ROWS * DIM;        // 9,633,792
  const int n_wq = QKV_DIM * DIM;     // 1,769,472
  const int n_wp = DIM * DIM;         //   589,824

  cvt_f32_bf16<<<(n_x  + 255) / 256, 256, 0, stream>>>(x,      x_bf,     n_x);
  cvt_f32_bf16<<<(n_wq + 255) / 256, 256, 0, stream>>>(w_qkv,  wqkv_bf,  n_wq);
  cvt_f32_bf16<<<(n_wp + 255) / 256, 256, 0, stream>>>(w_proj, wproj_bf, n_wp);

  // QKV: 392 x 72 wave-jobs of 32x32, 8 waves/block
  gemm_qkv<<<(ROWS / 32) * (QKV_DIM / 32) / 8, 256, 0, stream>>>(x_bf, wqkv_bf, qkv_bf);

  // Fused attention, one WG per (b,h), 314 KB dynamic LDS (CDNA5: 320 KB/WGP)
  (void)hipFuncSetAttribute((const void*)attn_core,
                            hipFuncAttributeMaxDynamicSharedMemorySize, SMEM_BYTES);
  attn_core<<<BATCH * HEADS, 512, SMEM_BYTES, stream>>>(qkv_bf, factors, o_bf);

  // Projection: 392 x 24 wave-jobs of 32x32, 8 waves/block
  gemm_proj<<<(ROWS / 32) * (DIM / 32) / 8, 256, 0, stream>>>(o_bf, wproj_bf, b_proj, out);
}
